// MultiheadAttention_16003048144898
// MI455X (gfx1250) — compile-verified, matrix operand order checked by reading, not run on hardware
//
#include <hip/hip_runtime.h>

typedef __bf16 bf16;
typedef __attribute__((ext_vector_type(16))) __bf16 v16bf;
typedef __attribute__((ext_vector_type(8)))  __bf16 v8bf;
typedef __attribute__((ext_vector_type(4)))  __bf16 v4bf;
typedef __attribute__((ext_vector_type(8)))  float  v8f;
typedef __attribute__((ext_vector_type(4)))  float  v4f;
typedef unsigned int u32;
typedef __attribute__((ext_vector_type(4))) u32 u32x4;
typedef __attribute__((ext_vector_type(8))) int i32x8;
typedef __attribute__((ext_vector_type(4))) int i32x4;

#define DEVI __device__ __forceinline__

// Problem constants
static constexpr int Bb = 4;
static constexpr int Ss = 2048;
static constexpr int Ee = 1024;
static constexpr int Hh = 16;
static constexpr int Dd = 64;

DEVI v16bf frag16(v8bf lo, v8bf hi) {
    union { v16bf v; v8bf h[2]; } u;
    u.h[0] = lo; u.h[1] = hi;
    return u.v;
}

DEVI v8f wmma_bf16(v16bf a, v16bf b, v8f c) {
    return __builtin_amdgcn_wmma_f32_16x16x32_bf16(false, a, false, b, (short)0, c,
                                                   false, false);
}

// ---------------------------------------------------------------------------
// Tensor Data Mover: 2D tile load Global -> LDS (D# per CDNA5 ISA)
//   data_size = 2 bytes (bf16). Optional LDS row padding.
//   pad_interval_code: pad after 2^(code+1) DWORDs; pad_amount_code: (code+1) DWORDs.
// ---------------------------------------------------------------------------
DEVI void tdm_load_2d(const void* gaddr, u32 lds_addr,
                      u32 tile_x, u32 tile_y,
                      u32 tensor_x, u32 tensor_y,
                      u32 row_stride,                 // elements between rows
                      u32 pad_interval_code, u32 pad_amount_code, int pad_enable) {
    const unsigned long long ga = (unsigned long long)gaddr;
    u32x4 g0;
    g0[0] = 1u;                                        // count=1 (valid user D#)
    g0[1] = lds_addr;                                  // LDS byte address
    g0[2] = (u32)(ga & 0xffffffffu);                   // global_addr[31:0]
    g0[3] = (u32)((ga >> 32) & 0x1ffffffu) | (2u << 30);  // ga[56:32] | type=2
    u32 w0 = (1u << 16);                               // data_size=1 -> 2 bytes
    if (pad_enable)
        w0 |= (1u << 20) | (pad_interval_code << 22) | (pad_amount_code << 25);
    i32x8 g1;
    g1[0] = (int)w0;
    g1[1] = (int)((tensor_x & 0xffffu) << 16);                         // dim0[15:0]
    g1[2] = (int)(((tensor_x >> 16) & 0xffffu) | ((tensor_y & 0xffffu) << 16));
    g1[3] = (int)(((tensor_y >> 16) & 0xffffu) | ((tile_x & 0xffffu) << 16));
    g1[4] = (int)(tile_y & 0xffffu);                   // tile_dim1 (tile_dim2 = 0)
    g1[5] = (int)row_stride;                           // dim0_stride[31:0]
    g1[6] = 0;                                         // dim0_stride[47:32], dim1_stride lo
    g1[7] = 0;
    const i32x4 z4 = {0, 0, 0, 0};
    const i32x8 z8 = {0, 0, 0, 0, 0, 0, 0, 0};
    __builtin_amdgcn_tensor_load_to_lds(g0, g1, z4, z4, z8, 0);
}

DEVI u32 lds_off(const void* p) {                      // flat->LDS byte offset
    return (u32)(unsigned long long)p;
}

// ---------------------------------------------------------------------------
// Kernel 1: fp32 -> bf16 elementwise convert
// ---------------------------------------------------------------------------
__global__ __launch_bounds__(256)
void k_cvt_bf16(const float* __restrict__ in, bf16* __restrict__ out, int n) {
    int i = (blockIdx.x * blockDim.x + threadIdx.x) * 4;
    if (i < n) {
        v4f x = *(const v4f*)(in + i);
        v4bf y;
        y[0] = (bf16)x[0]; y[1] = (bf16)x[1]; y[2] = (bf16)x[2]; y[3] = (bf16)x[3];
        *(v4bf*)(out + i) = y;
    }
}

// ---------------------------------------------------------------------------
// Kernel 2: fp32 [K][N] -> bf16 [N][K] (transpose + convert via LDS tile)
// ---------------------------------------------------------------------------
__global__ __launch_bounds__(256)
void k_transpose_cvt(const float* __restrict__ in, bf16* __restrict__ out,
                     int K, int N) {
    __shared__ float tile[32][33];
    const int n0 = blockIdx.x * 32;
    const int k0 = blockIdx.y * 32;
    const int tx = threadIdx.x;
    const int ty = threadIdx.y;
#pragma unroll
    for (int i = 0; i < 4; ++i)
        tile[ty + 8 * i][tx] = in[(size_t)(k0 + ty + 8 * i) * N + (n0 + tx)];
    __syncthreads();
#pragma unroll
    for (int i = 0; i < 4; ++i)
        out[(size_t)(n0 + ty + 8 * i) * K + (k0 + tx)] = (bf16)tile[tx][ty + 8 * i];
}

// ---------------------------------------------------------------------------
// Kernel 3/5: bf16 GEMM  C[M,N] = A[M,K] * Bt[N,K]^T + bias[N]
//   Tiles staged by the Tensor Data Mover, double-buffered LDS.
//   MODE 0: scatter into Q/K/V bf16 [B,H,S,D]; Q additionally scaled 1/sqrt(D)
//   MODE 1: fp32 row-major [M][N] output
// ---------------------------------------------------------------------------
template <int MODE>
__global__ __launch_bounds__(256)
void k_gemm_bf16(const bf16* __restrict__ A, const bf16* __restrict__ Bt,
                 const float* __restrict__ bias, int N, int K,
                 float* __restrict__ outF, bf16* __restrict__ outQKV) {
    // 32-elem rows padded to 40 (TDM pad: 16 DWORDs -> code 3, +4 DWORDs -> code 3)
    __shared__ __align__(16) bf16 As[2][128][40];
    __shared__ __align__(16) bf16 Bs[2][128][40];

    const int tid  = threadIdx.x;
    const int lane = tid & 31;
    const int wid  = tid >> 5;
    const int wr   = wid >> 2;
    const int wc   = wid & 3;
    const int ln   = lane & 15;
    const int half = lane >> 4;

    const int bm = blockIdx.y * 128;
    const int bn = blockIdx.x * 128;

    const v8f zero8 = {0.f, 0.f, 0.f, 0.f, 0.f, 0.f, 0.f, 0.f};
    v8f acc[4][2];
#pragma unroll
    for (int i = 0; i < 4; ++i)
#pragma unroll
        for (int j = 0; j < 2; ++j) acc[i][j] = zero8;

    const int KT = K >> 5;
    if (wid == 0) {
        tdm_load_2d(A + (size_t)bm * K, lds_off(&As[0][0][0]), 32, 128,
                    (u32)K, 1u << 20, (u32)K, 3, 3, 1);
        tdm_load_2d(Bt + (size_t)bn * K, lds_off(&Bs[0][0][0]), 32, 128,
                    (u32)K, 1u << 20, (u32)K, 3, 3, 1);
    }

    int cur = 0;
    for (int kt = 0; kt < KT; ++kt) {
        if (wid == 0 && kt + 1 < KT) {
            const int nb = cur ^ 1;
            tdm_load_2d(A + (size_t)bm * K + (kt + 1) * 32, lds_off(&As[nb][0][0]),
                        32, 128, (u32)K, 1u << 20, (u32)K, 3, 3, 1);
            tdm_load_2d(Bt + (size_t)bn * K + (kt + 1) * 32, lds_off(&Bs[nb][0][0]),
                        32, 128, (u32)K, 1u << 20, (u32)K, 3, 3, 1);
        }
        if (kt + 1 < KT) __builtin_amdgcn_s_wait_tensorcnt(2);
        else             __builtin_amdgcn_s_wait_tensorcnt(0);
        __syncthreads();

        v16bf af[4], bfr[2];
#pragma unroll
        for (int i = 0; i < 4; ++i) {
            const bf16* p = &As[cur][wr * 64 + i * 16 + ln][0];
            af[i] = frag16(*(const v8bf*)(p + 8 * half),
                           *(const v8bf*)(p + 16 + 8 * half));
        }
#pragma unroll
        for (int j = 0; j < 2; ++j) {
            const bf16* p = &Bs[cur][wc * 32 + j * 16 + ln][0];
            bfr[j] = frag16(*(const v8bf*)(p + 8 * half),
                            *(const v8bf*)(p + 16 + 8 * half));
        }
#pragma unroll
        for (int i = 0; i < 4; ++i)
#pragma unroll
            for (int j = 0; j < 2; ++j)
                acc[i][j] = wmma_bf16(af[i], bfr[j], acc[i][j]);

        __syncthreads();
        cur ^= 1;
    }

    // Epilogue. C layout: lane ln = column, VGPR v = row (v + 8*half).
#pragma unroll
    for (int i = 0; i < 4; ++i) {
#pragma unroll
        for (int j = 0; j < 2; ++j) {
            const int   n     = bn + wc * 32 + j * 16 + ln;
            const float bv    = bias[n];
            const int   mbase = bm + wr * 64 + i * 16 + 8 * half;
            if (MODE == 1) {
#pragma unroll
                for (int v = 0; v < 8; ++v)
                    outF[(size_t)(mbase + v) * N + n] = acc[i][j][v] + bv;
            } else {
                const int which = n >> 10;       // 0=Q,1=K,2=V
                const int nn    = n & 1023;
                const int h     = nn >> 6;
                const int d     = nn & 63;
                const float qs  = (which == 0) ? 0.125f : 1.0f;  // fold 1/sqrt(D)
                bf16* dst = outQKV + (size_t)which * ((size_t)Bb * Hh * Ss * Dd);
#pragma unroll
                for (int v = 0; v < 8; ++v) {
                    const int m = mbase + v;
                    const int b = m >> 11;
                    const int s = m & 2047;
                    dst[(((size_t)(b * Hh + h)) * Ss + s) * Dd + d] =
                        (bf16)((acc[i][j][v] + bv) * qs);
                }
            }
        }
    }
}

// ---------------------------------------------------------------------------
// Kernel 4: flash attention. Block = (128 q rows, head, batch), 8 waves.
// K tiles via TDM (double-buffered), V transposed manually, P via LDS staging.
// ---------------------------------------------------------------------------
__global__ __launch_bounds__(256)
void k_flash_attn(const bf16* __restrict__ Qb, const bf16* __restrict__ Kb,
                  const bf16* __restrict__ Vb, bf16* __restrict__ Ob) {
    // 64-elem rows padded to 72 (TDM pad: 32 DWORDs -> code 4, +4 DWORDs -> code 3)
    __shared__ __align__(16) bf16 Ks[2][64][72];
    __shared__ __align__(16) bf16 Vt[2][64][72];
    __shared__ __align__(16) bf16 Ps[8][16][72];

    const int tid  = threadIdx.x;
    const int lane = tid & 31;
    const int wid  = tid >> 5;
    const int ln   = lane & 15;
    const int half = lane >> 4;

    const int qblk = blockIdx.x;
    const int h    = blockIdx.y;
    const int b    = blockIdx.z;

    const size_t headoff = ((size_t)(b * Hh + h)) * Ss * Dd;
    const bf16* Q  = Qb + headoff;
    const bf16* Kp = Kb + headoff;
    const bf16* Vp = Vb + headoff;

    // Q fragments (A layout; Q pre-scaled by 1/sqrt(D) in the QKV epilogue)
    const int   qrow = qblk * 128 + wid * 16 + ln;
    const bf16* qp   = Q + (size_t)qrow * Dd;
    v16bf qf0 = frag16(*(const v8bf*)(qp + 8 * half),
                       *(const v8bf*)(qp + 16 + 8 * half));
    v16bf qf1 = frag16(*(const v8bf*)(qp + 32 + 8 * half),
                       *(const v8bf*)(qp + 48 + 8 * half));

    const v8f zero8 = {0.f, 0.f, 0.f, 0.f, 0.f, 0.f, 0.f, 0.f};
    v8f o[4];
#pragma unroll
    for (int dt = 0; dt < 4; ++dt) o[dt] = zero8;
    float mrow[8], lrow[8];
#pragma unroll
    for (int v = 0; v < 8; ++v) { mrow[v] = -1e30f; lrow[v] = 0.f; }

    // V staging: 4 threads per 64-key row, 16 elems (32B) each
    const int skr = tid >> 2;
    const int sco = (tid & 3) * 16;

    if (wid == 0)
        tdm_load_2d(Kp, lds_off(&Ks[0][0][0]), 64, 64, 64, (u32)Ss, 64, 4, 3, 1);
    v8bf rv0 = *(const v8bf*)(Vp + (size_t)skr * Dd + sco);
    v8bf rv1 = *(const v8bf*)(Vp + (size_t)skr * Dd + sco + 8);
#pragma unroll
    for (int e = 0; e < 8; ++e) {
        Vt[0][sco + e][skr]     = rv0[e];
        Vt[0][sco + 8 + e][skr] = rv1[e];
    }

    const int NIT = Ss / 64;
    int cur = 0;
    for (int it = 0; it < NIT; ++it) {
        if (it + 1 < NIT) {
            if (wid == 0)
                tdm_load_2d(Kp + (size_t)(it + 1) * 64 * Dd, lds_off(&Ks[cur ^ 1][0][0]),
                            64, 64, 64, (u32)Ss, 64, 4, 3, 1);
            const bf16* vp2 = Vp + (size_t)((it + 1) * 64 + skr) * Dd + sco;
            rv0 = *(const v8bf*)(vp2);
            rv1 = *(const v8bf*)(vp2 + 8);
        }
        if (it + 1 < NIT) __builtin_amdgcn_s_wait_tensorcnt(1);
        else              __builtin_amdgcn_s_wait_tensorcnt(0);
        __syncthreads();

        // S = Q * K^T (4 tiles of 16 keys)
        v8f sc[4];
#pragma unroll
        for (int j = 0; j < 4; ++j) {
            const bf16* kb = &Ks[cur][j * 16 + ln][0];
            v16bf b0 = frag16(*(const v8bf*)(kb + 8 * half),
                              *(const v8bf*)(kb + 16 + 8 * half));
            v16bf b1 = frag16(*(const v8bf*)(kb + 32 + 8 * half),
                              *(const v8bf*)(kb + 48 + 8 * half));
            v8f z = zero8;
            z = wmma_bf16(qf0, b0, z);
            z = wmma_bf16(qf1, b1, z);
            sc[j] = z;
        }

        // online softmax (row = v + 8*half, reduce across 16 lanes of the half)
        float rm[8];
#pragma unroll
        for (int v = 0; v < 8; ++v)
            rm[v] = fmaxf(fmaxf(sc[0][v], sc[1][v]), fmaxf(sc[2][v], sc[3][v]));
#pragma unroll
        for (int off = 1; off < 16; off <<= 1)
#pragma unroll
            for (int v = 0; v < 8; ++v)
                rm[v] = fmaxf(rm[v], __shfl_xor(rm[v], off, 32));

        float al[8];
#pragma unroll
        for (int v = 0; v < 8; ++v) {
            const float mn = fmaxf(mrow[v], rm[v]);
            al[v]   = __expf(mrow[v] - mn);
            mrow[v] = mn;
        }
#pragma unroll
        for (int j = 0; j < 4; ++j)
#pragma unroll
            for (int v = 0; v < 8; ++v) sc[j][v] = __expf(sc[j][v] - mrow[v]);

        float rs[8];
#pragma unroll
        for (int v = 0; v < 8; ++v)
            rs[v] = sc[0][v] + sc[1][v] + sc[2][v] + sc[3][v];
#pragma unroll
        for (int off = 1; off < 16; off <<= 1)
#pragma unroll
            for (int v = 0; v < 8; ++v) rs[v] += __shfl_xor(rs[v], off, 32);
#pragma unroll
        for (int v = 0; v < 8; ++v) lrow[v] = lrow[v] * al[v] + rs[v];
#pragma unroll
        for (int dt = 0; dt < 4; ++dt)
#pragma unroll
            for (int v = 0; v < 8; ++v) o[dt][v] *= al[v];

        // transpose P (C layout) -> A layout via per-wave LDS staging
#pragma unroll
        for (int j = 0; j < 4; ++j)
#pragma unroll
            for (int v = 0; v < 8; ++v)
                Ps[wid][v + 8 * half][j * 16 + ln] = (bf16)sc[j][v];

        const bf16* pp = &Ps[wid][ln][0];
        v16bf pa0 = frag16(*(const v8bf*)(pp + 8 * half),
                           *(const v8bf*)(pp + 16 + 8 * half));
        v16bf pa1 = frag16(*(const v8bf*)(pp + 32 + 8 * half),
                           *(const v8bf*)(pp + 48 + 8 * half));

        // O += P * V
#pragma unroll
        for (int dt = 0; dt < 4; ++dt) {
            const bf16* vb2 = &Vt[cur][dt * 16 + ln][0];
            v16bf b0 = frag16(*(const v8bf*)(vb2 + 8 * half),
                              *(const v8bf*)(vb2 + 16 + 8 * half));
            v16bf b1 = frag16(*(const v8bf*)(vb2 + 32 + 8 * half),
                              *(const v8bf*)(vb2 + 48 + 8 * half));
            o[dt] = wmma_bf16(pa0, b0, o[dt]);
            o[dt] = wmma_bf16(pa1, b1, o[dt]);
        }

        if (it + 1 < NIT) {
#pragma unroll
            for (int e = 0; e < 8; ++e) {
                Vt[cur ^ 1][sco + e][skr]     = rv0[e];
                Vt[cur ^ 1][sco + 8 + e][skr] = rv1[e];
            }
        }
        __syncthreads();
        cur ^= 1;
    }

    // normalize + write O as bf16 [B*S][E]
    float inv[8];
#pragma unroll
    for (int v = 0; v < 8; ++v) inv[v] = 1.0f / lrow[v];
    const int srow0 = qblk * 128 + wid * 16 + 8 * half;
#pragma unroll
    for (int dt = 0; dt < 4; ++dt) {
        const int d = h * 64 + dt * 16 + ln;
#pragma unroll
        for (int v = 0; v < 8; ++v) {
            const int srow = srow0 + v;
            Ob[(size_t)(b * Ss + srow) * Ee + d] = (bf16)(o[dt][v] * inv[v]);
        }
    }
}

// ---------------------------------------------------------------------------
// Host launch
// ---------------------------------------------------------------------------
extern "C" void kernel_launch(void* const* d_in, const int* in_sizes, int n_in,
                              void* d_out, int out_size, void* d_ws, size_t ws_size,
                              hipStream_t stream) {
    (void)in_sizes; (void)n_in; (void)out_size; (void)ws_size;
    const float* x     = (const float*)d_in[0];
    const float* w_qkv = (const float*)d_in[1];
    const float* b_qkv = (const float*)d_in[2];
    const float* w_out = (const float*)d_in[3];
    const float* b_out = (const float*)d_in[4];
    float* out = (float*)d_out;

    char*  ws  = (char*)d_ws;
    size_t off = 0;
    auto carve = [&](size_t bytes) -> char* {
        char* p = ws + off;
        off += (bytes + 255) & ~(size_t)255;
        return p;
    };
    const size_t nBS   = (size_t)Bb * Ss;
    const size_t headN = (size_t)Bb * Hh * Ss * Dd;

    bf16* xb    = (bf16*)carve(nBS * Ee * 2);
    bf16* wqkvt = (bf16*)carve((size_t)3 * Ee * Ee * 2);
    bf16* woutt = (bf16*)carve((size_t)Ee * Ee * 2);
    bf16* qkvb  = (bf16*)carve(3 * headN * 2);
    bf16* ob    = (bf16*)carve(nBS * Ee * 2);

    k_cvt_bf16<<<(int)(nBS * Ee / (256 * 4)), 256, 0, stream>>>(x, xb, (int)(nBS * Ee));
    k_transpose_cvt<<<dim3(3072 / 32, 1024 / 32), dim3(32, 8), 0, stream>>>(
        w_qkv, wqkvt, 1024, 3072);
    k_transpose_cvt<<<dim3(1024 / 32, 1024 / 32), dim3(32, 8), 0, stream>>>(
        w_out, woutt, 1024, 1024);
    k_gemm_bf16<0><<<dim3(3072 / 128, 8192 / 128), 256, 0, stream>>>(
        xb, wqkvt, b_qkv, 3072, 1024, nullptr, qkvb);
    k_flash_attn<<<dim3(Ss / 128, Hh, Bb), 256, 0, stream>>>(
        qkvb, qkvb + headN, qkvb + 2 * headN, ob);
    k_gemm_bf16<1><<<dim3(1024 / 128, 8192 / 128), 256, 0, stream>>>(
        ob, woutt, b_out, 1024, 1024, out, nullptr);
}
